// SpiralNet_17995912970891
// MI455X (gfx1250) — matrix operand order repaired and checked by reading, not compiled, not run on hardware
//
#include <hip/hip_runtime.h>
#include <hip/hip_bf16.h>
#include <math.h>

// ---------------------------------------------------------------------------
// CDNA5 (gfx1250) SpiralNet forward.  All GEMM-shaped work goes through
// v_wmma_f32_16x16x32_f16 (wave32 WMMA).  fp32 activations live in the 192MB
// L2; weights are pre-packed to f16 [N][K] so B-tile staging is a pure b128
// copy; A tiles are converted f32->f16 while staging into LDS.
// ---------------------------------------------------------------------------

typedef __attribute__((ext_vector_type(16))) _Float16 v16h;
typedef __attribute__((ext_vector_type(8)))  float    v8f;

union Frag16 { uint4 q[2]; v16h h; };

__device__ __forceinline__ float gelu_f(float x) {
  return 0.5f * x * (1.0f + erff(x * 0.70710678118654752440f));
}

// ---------------------------------------------------------------------------
// Weight repack: out[n*Kpad + k] = f16( wNK ? W[n][k] : W[k][n] ), zero-padded
// to Kpad (multiple of 32).  Launched immediately before the GEMM that uses it.
// ---------------------------------------------------------------------------
__global__ __launch_bounds__(256)
void convert_w_k(const float* __restrict__ W, int wNK, int N, int Kw, int Kpad,
                 _Float16* __restrict__ out)
{
  size_t idx = (size_t)blockIdx.x * 256u + threadIdx.x;
  size_t total = (size_t)N * Kpad;
  if (idx >= total) return;
  int k = (int)(idx % Kpad);
  int n = (int)(idx / Kpad);
  float v = 0.f;
  if (k < Kw) v = wNK ? W[(size_t)n * Kw + k] : W[(size_t)k * N + n];
  out[idx] = (_Float16)v;
}

// ---------------------------------------------------------------------------
// WMMA GEMM:  out[M,N] = A[M,K] * Wh[N,K]^T + bias (+gelu) (+res1) (+res2)
//   A  : fp32 [M][K], K multiple of 32
//   Wh : f16  [N][K] packed by convert_w_k
// Block tile 128x64, 256 threads = 8 waves; each wave owns a 32x32 patch:
// 2 A-frags x 2 B-frags -> 4 v_wmma_f32_16x16x32_f16 per K-step.
// Staging is branchless (clamped rows + select-zero) so all global loads for
// one K-step issue under a single wait.
// ---------------------------------------------------------------------------
__global__ __launch_bounds__(256)
void gemm_wmma(const float* __restrict__ A, int M, int K,
               const _Float16* __restrict__ Wh,
               const float* __restrict__ bias, int N,
               const float* res1, const float* res2,
               float* out, int act)
{
  __shared__ alignas(16) _Float16 lA[128][40];  // 40-half stride: 80B rows
  __shared__ alignas(16) _Float16 lB[64][40];   // lB[n][k]

  const int tid  = threadIdx.x;
  const int lane = tid & 31;
  const int wv   = tid >> 5;
  const int m0   = blockIdx.x * 128;
  const int n0   = blockIdx.y * 64;
  const int wr   = (wv & 3) * 32;     // wave row offset within tile
  const int wc   = (wv >> 2) * 32;    // wave col offset within tile
  const int r    = lane & 15;
  const int hi   = lane >> 4;

  // per-thread staging coordinates (branchless: clamp row, zero via select)
  int arow[4], acol[4]; bool aval[4];
#pragma unroll
  for (int it = 0; it < 4; ++it) {
    int idx  = tid + it * 256;           // 1024 float4 slots (128 x 8)
    int row  = idx >> 3;
    acol[it] = (idx & 7) * 4;
    aval[it] = (m0 + row) < M;
    arow[it] = aval[it] ? (m0 + row) : (M - 1);
  }
  const int bn   = tid >> 2;             // 0..63
  const int bc4  = (tid & 3) * 8;        // half offset of 16B chunk
  const bool bvl = (n0 + bn) < N;
  const int brow = bvl ? (n0 + bn) : (N - 1);

  v8f acc00 = {}, acc01 = {}, acc10 = {}, acc11 = {};

  for (int k0 = 0; k0 < K; k0 += 32) {
    // ---- issue all global loads for this K-step back-to-back
    float4 va[4];
#pragma unroll
    for (int it = 0; it < 4; ++it)
      va[it] = *reinterpret_cast<const float4*>(
                   A + (size_t)arow[it] * K + (k0 + acol[it]));
    uint4 vb = *reinterpret_cast<const uint4*>(
                   Wh + (size_t)brow * K + (k0 + bc4));
    if (k0 + 32 < K) {                   // uniform: prefetch next K-slice
#pragma unroll
      for (int it = 0; it < 4; ++it)
        __builtin_prefetch(A + (size_t)arow[it] * K + (k0 + 32 + acol[it]), 0, 1);
    }
    // ---- convert + store to LDS (zero out-of-range rows)
#pragma unroll
    for (int it = 0; it < 4; ++it) {
      float4 v = va[it];
      if (!aval[it]) v = make_float4(0.f, 0.f, 0.f, 0.f);
      int row = (tid + it * 256) >> 3;
      lA[row][acol[it] + 0] = (_Float16)v.x;
      lA[row][acol[it] + 1] = (_Float16)v.y;
      lA[row][acol[it] + 2] = (_Float16)v.z;
      lA[row][acol[it] + 3] = (_Float16)v.w;
    }
    if (!bvl) vb = make_uint4(0u, 0u, 0u, 0u);
    *reinterpret_cast<uint4*>(&lB[bn][bc4]) = vb;
    __syncthreads();

    // ---- fragments (16B LDS reads, layouts per CDNA5 ISA 7.12.2)
    Frag16 fa0, fa1, fb0, fb1;
    const uint4* rowA0 = reinterpret_cast<const uint4*>(&lA[wr + r][0]);
    const uint4* rowA1 = reinterpret_cast<const uint4*>(&lA[wr + 16 + r][0]);
    fa0.q[0] = rowA0[hi];     fa0.q[1] = rowA0[2 + hi];
    fa1.q[0] = rowA1[hi];     fa1.q[1] = rowA1[2 + hi];
    const uint4* rowB0 = reinterpret_cast<const uint4*>(&lB[wc + r][0]);
    const uint4* rowB1 = reinterpret_cast<const uint4*>(&lB[wc + 16 + r][0]);
    fb0.q[0] = rowB0[hi];     fb0.q[1] = rowB0[2 + hi];
    fb1.q[0] = rowB1[hi];     fb1.q[1] = rowB1[2 + hi];

    acc00 = __builtin_amdgcn_wmma_f32_16x16x32_f16(false, fa0.h, false, fb0.h,
                                                   (short)0, acc00, false, false);
    acc01 = __builtin_amdgcn_wmma_f32_16x16x32_f16(false, fa0.h, false, fb1.h,
                                                   (short)0, acc01, false, false);
    acc10 = __builtin_amdgcn_wmma_f32_16x16x32_f16(false, fa1.h, false, fb0.h,
                                                   (short)0, acc10, false, false);
    acc11 = __builtin_amdgcn_wmma_f32_16x16x32_f16(false, fa1.h, false, fb1.h,
                                                   (short)0, acc11, false, false);
    __syncthreads();
  }

  // ---- epilogue: lane = col (L&15), VGPR vr -> row vr + (L>>4)*8
  const int gn0 = n0 + wc + r;
  const int gn1 = gn0 + 16;
  const float b0 = (gn0 < N) ? bias[gn0] : 0.f;
  const float b1 = (gn1 < N) ? bias[gn1] : 0.f;
#pragma unroll
  for (int mi = 0; mi < 2; ++mi) {
    const v8f& a0 = mi ? acc10 : acc00;
    const v8f& a1 = mi ? acc11 : acc01;
#pragma unroll
    for (int vr = 0; vr < 8; ++vr) {
      int gm = m0 + wr + mi * 16 + hi * 8 + vr;
      if (gm >= M) continue;
      size_t ro = (size_t)gm * N;
      if (gn0 < N) {
        float v = a0[vr] + b0;
        if (act) v = gelu_f(v);
        if (res1) v += res1[ro + gn0];
        if (res2) v += res2[ro + gn0];
        out[ro + gn0] = v;
      }
      if (gn1 < N) {
        float v = a1[vr] + b1;
        if (act) v = gelu_f(v);
        if (res1) v += res1[ro + gn1];
        if (res2) v += res2[ro + gn1];
        out[ro + gn1] = v;
      }
    }
  }
}

// ---------------------------------------------------------------------------
// LayerNorm over last dim (one wave32 per token)
// ---------------------------------------------------------------------------
__global__ __launch_bounds__(256)
void layernorm_k(const float* __restrict__ x, const float* __restrict__ g,
                 const float* __restrict__ b, float* __restrict__ y,
                 int M, int C)
{
  int wv = threadIdx.x >> 5, lane = threadIdx.x & 31;
  int t = blockIdx.x * 8 + wv;
  if (t >= M) return;
  const float* xr = x + (size_t)t * C;
  float s = 0.f, ss = 0.f;
  for (int c = lane; c < C; c += 32) { float v = xr[c]; s += v; ss += v * v; }
#pragma unroll
  for (int o = 16; o; o >>= 1) {
    s  += __shfl_xor(s, o, 32);
    ss += __shfl_xor(ss, o, 32);
  }
  float mean = s / C;
  float var  = ss / C - mean * mean;
  float rs   = rsqrtf(var + 1e-5f);
  float* yr = y + (size_t)t * C;
  for (int c = lane; c < C; c += 32)
    yr[c] = (xr[c] - mean) * rs * g[c] + b[c];
}

// ---------------------------------------------------------------------------
// Spiral gather: Ag[m, c] = xn[b, h+dy(c), w+dx(c), c] (zero outside)
// ---------------------------------------------------------------------------
__global__ __launch_bounds__(256)
void spiral_gather_k(const float* __restrict__ xn, float* __restrict__ Ag,
                     int B, int H, int Wd, int C)
{
  size_t idx = (size_t)blockIdx.x * 256u + threadIdx.x;
  size_t total = (size_t)B * H * Wd * C;
  if (idx >= total) return;
  int c = (int)(idx % C);
  size_t m = idx / C;
  int w = (int)(m % Wd);
  int h = (int)((m / Wd) % H);
  int b = (int)(m / ((size_t)H * Wd));
  int half = C >> 1;
  int i = c % half;
  float rf  = 14.0f / (float)half;
  float ang = 3.14159265358979323846f * (float)i / 16.0f;
  float mag = (i <= (half >> 1)) ? ((float)i * rf) : (14.0f - (float)i * rf);
  int dy = (int)rintf(mag * cosf(ang));   // rintf == Python round (RNE)
  int dx = (int)rintf(mag * sinf(ang));
  int ih = h + dy, iw = w + dx;
  float v = 0.f;
  if (ih >= 0 && ih < H && iw >= 0 && iw < Wd)
    v = xn[(((size_t)b * H + ih) * Wd + iw) * C + c];
  Ag[idx] = v;
}

// ---------------------------------------------------------------------------
// im2col for the 7x7 s4 p2 patch conv; K padded 147 -> 160
// ---------------------------------------------------------------------------
__global__ __launch_bounds__(256)
void im2col_patch_k(const float* __restrict__ img, float* __restrict__ Ag, int B)
{
  size_t idx = (size_t)blockIdx.x * 256u + threadIdx.x;
  size_t total = (size_t)B * 3136 * 160;
  if (idx >= total) return;
  int k = (int)(idx % 160);
  size_t m = idx / 160;
  float v = 0.f;
  if (k < 147) {
    int ow = (int)(m % 56);
    int oh = (int)((m / 56) % 56);
    int b  = (int)(m / 3136);
    int ci = k / 49, rr = k % 49, kh = rr / 7, kw = rr % 7;
    int ih = oh * 4 - 2 + kh;
    int iw = ow * 4 - 2 + kw;
    if (ih >= 0 && ih < 224 && iw >= 0 && iw < 224)
      v = img[(((size_t)b * 3 + ci) * 224 + ih) * 224 + iw];
  }
  Ag[idx] = v;
}

// ---------------------------------------------------------------------------
// im2col for the 3x3 s2 p1 downsample convs (NHWC input, k = c*9+kh*3+kw)
// ---------------------------------------------------------------------------
__global__ __launch_bounds__(256)
void im2col3x3_k(const float* __restrict__ x, float* __restrict__ Ag,
                 int B, int Hin, int Win, int C, int Hout, int Wout)
{
  size_t K = (size_t)C * 9;
  size_t total = (size_t)B * Hout * Wout * K;
  size_t idx = (size_t)blockIdx.x * 256u + threadIdx.x;
  if (idx >= total) return;
  int k = (int)(idx % K);
  size_t m = idx / K;
  int c = k / 9, rr = k % 9, kh = rr / 3, kw = rr % 3;
  int ow = (int)(m % Wout);
  int oh = (int)((m / Wout) % Hout);
  int b  = (int)(m / ((size_t)Hout * Wout));
  int ih = oh * 2 - 1 + kh;
  int iw = ow * 2 - 1 + kw;
  float v = 0.f;
  if (ih >= 0 && ih < Hin && iw >= 0 && iw < Win)
    v = x[(((size_t)b * Hin + ih) * Win + iw) * C + c];
  Ag[idx] = v;
}

// ---------------------------------------------------------------------------
// out[b,c] = mean over HW of (a [+ b2])
// ---------------------------------------------------------------------------
__global__ __launch_bounds__(256)
void pool_mean_k(const float* __restrict__ a, const float* __restrict__ b2,
                 float* __restrict__ out, int B, int HW, int C)
{
  int idx = blockIdx.x * 256 + threadIdx.x;
  if (idx >= B * C) return;
  int c = idx % C, b = idx / C;
  const float* pa = a + (size_t)b * HW * C + c;
  float s = 0.f;
  if (b2) {
    const float* pb = b2 + (size_t)b * HW * C + c;
    for (int hw = 0; hw < HW; ++hw) s += pa[(size_t)hw * C] + pb[(size_t)hw * C];
  } else {
    for (int hw = 0; hw < HW; ++hw) s += pa[(size_t)hw * C];
  }
  out[idx] = s / (float)HW;
}

// ---------------------------------------------------------------------------
// Tiny channel-weight MLP + pair softmax (one block per batch row)
// ---------------------------------------------------------------------------
__global__ __launch_bounds__(256)
void channel_mlp_k(const float* __restrict__ w,
                   const float* __restrict__ rw1w, const float* __restrict__ rw1b,
                   const float* __restrict__ rw2w, const float* __restrict__ rw2b,
                   float* __restrict__ Aout, float* __restrict__ Sout, int C)
{
  __shared__ float ws[512];
  __shared__ float hs[128];
  int b = blockIdx.x, tid = threadIdx.x;
  int C4 = C >> 2, C2 = C * 2;
  for (int c = tid; c < C; c += 256) ws[c] = w[(size_t)b * C + c];
  __syncthreads();
  for (int j = tid; j < C4; j += 256) {
    float acc = rw1b[j];
    for (int k = 0; k < C; ++k) acc += ws[k] * rw1w[(size_t)k * C4 + j];
    hs[j] = gelu_f(acc);
  }
  __syncthreads();
  for (int c = tid; c < C; c += 256) {
    float v0 = rw2b[2 * c], v1 = rw2b[2 * c + 1];
    for (int k = 0; k < C4; ++k) {
      float h = hs[k];
      v0 += h * rw2w[(size_t)k * C2 + 2 * c];
      v1 += h * rw2w[(size_t)k * C2 + 2 * c + 1];
    }
    float mx = fmaxf(v0, v1);
    float e0 = expf(v0 - mx), e1 = expf(v1 - mx);
    float inv = 1.0f / (e0 + e1);
    Aout[(size_t)b * C + c] = e0 * inv;
    Sout[(size_t)b * C + c] = e1 * inv;
  }
}

// ---------------------------------------------------------------------------
// Gating combine (in-place into xc): xc = xc*a[b,c] + xs*s[b,c]
// ---------------------------------------------------------------------------
__global__ __launch_bounds__(256)
void combine_k(float* __restrict__ xc, const float* __restrict__ xs,
               const float* __restrict__ Aw, const float* __restrict__ Sw,
               int HW, int C, size_t total)
{
  size_t idx = (size_t)blockIdx.x * 256u + threadIdx.x;
  if (idx >= total) return;
  int c = (int)(idx % C);
  size_t m = idx / C;
  int b = (int)(m / HW);
  xc[idx] = xc[idx] * Aw[(size_t)b * C + c] + xs[idx] * Sw[(size_t)b * C + c];
}

// ---------------------------------------------------------------------------
// Host orchestration
// ---------------------------------------------------------------------------
struct BlockP {
  const float *ln1_g, *ln1_b, *ln2_g, *ln2_b, *self_w, *self_b, *sp_w, *sp_b,
              *rw1_w, *rw1_b, *rw2_w, *rw2_b, *proj_w, *proj_b,
              *m1_w, *m1_b, *m2_w, *m2_b;
};

extern "C" void kernel_launch(void* const* d_in, const int* in_sizes, int n_in,
                              void* d_out, int out_size, void* d_ws, size_t ws_size,
                              hipStream_t stream)
{
  (void)in_sizes; (void)n_in; (void)out_size; (void)ws_size;
  static const int DIMS[4]   = {64, 128, 320, 512};
  static const int LAYERS[4] = {2, 2, 4, 2};
  const int Bn = 32;

  const float* img     = (const float*)d_in[0];
  const float* patch_w = (const float*)d_in[1];
  const float* patch_b = (const float*)d_in[2];
  int cur = 3;
  BlockP blocks[4][4];
  const float* down_w[3]; const float* down_b[3];
  for (int s = 0; s < 4; ++s) {
    for (int l = 0; l < LAYERS[s]; ++l) {
      const float** p = (const float**)&blocks[s][l];
      for (int t = 0; t < 18; ++t) p[t] = (const float*)d_in[cur++];
    }
    if (s < 3) {
      down_w[s] = (const float*)d_in[cur++];
      down_b[s] = (const float*)d_in[cur++];
    }
  }
  const float* norm_g = (const float*)d_in[cur++];
  const float* norm_b = (const float*)d_in[cur++];
  const float* head_w = (const float*)d_in[cur++];
  const float* head_b = (const float*)d_in[cur++];

  // workspace arena
  char* ws = (char*)d_ws;
  const size_t ACT   = (size_t)32 * 56 * 56 * 64 * 4;    // 25.7 MB
  const size_t BIGSZ = (size_t)32 * 56 * 56 * 256 * 4;   // 102.8 MB
  float* X   = (float*)(ws);
  float* XN  = (float*)(ws + ACT);
  float* XC  = (float*)(ws + 2 * ACT);
  float* XS  = (float*)(ws + 3 * ACT);
  float* BIG = (float*)(ws + 4 * ACT);
  float* Wp  = (float*)(ws + 4 * ACT + BIGSZ);
  float* Aw  = Wp + 32 * 512;
  float* Sw  = Aw + 32 * 512;
  _Float16* WH16 = (_Float16*)(Sw + 32 * 512);           // 4MB repacked weight

  auto cdiv = [](size_t a, size_t b) { return (unsigned)((a + b - 1) / b); };
  // repack weight to f16 [N][Kpad] then run WMMA GEMM (stream-serialized)
  auto gemm = [&](const float* A, int M, int K, int Kw, const float* W, int wNK,
                  const float* bias, int N, const float* r1, const float* r2,
                  float* out, int act) {
    convert_w_k<<<cdiv((size_t)N * K, 256), 256, 0, stream>>>(W, wNK, N, Kw, K, WH16);
    dim3 g(cdiv(M, 128), cdiv(N, 64));
    gemm_wmma<<<g, 256, 0, stream>>>(A, M, K, WH16, bias, N, r1, r2, out, act);
  };

  // ---- patch embed: im2col + WMMA GEMM -> X (NHWC, 32x56x56x64)
  {
    size_t tot = (size_t)Bn * 3136 * 160;
    im2col_patch_k<<<cdiv(tot, 256), 256, 0, stream>>>(img, BIG, Bn);
    gemm(BIG, Bn * 3136, 160, 147, patch_w, 1, patch_b, 64, nullptr, nullptr, X, 0);
  }

  int H = 56;
  for (int s = 0; s < 4; ++s) {
    const int C = DIMS[s];
    const int HW = H * H;
    const int M = Bn * HW;
    for (int l = 0; l < LAYERS[s]; ++l) {
      const BlockP& bp = blocks[s][l];
      // LN1 -> XN
      layernorm_k<<<cdiv(M, 8), 256, 0, stream>>>(X, bp.ln1_g, bp.ln1_b, XN, M, C);
      // spiral FC: gather -> BIG, GEMM -> XC   (sp_w is [O][C] = [N][K])
      spiral_gather_k<<<cdiv((size_t)M * C, 256), 256, 0, stream>>>(XN, BIG, Bn, H, H, C);
      gemm(BIG, M, C, C, bp.sp_w, 1, bp.sp_b, C, nullptr, nullptr, XC, 0);
      // self FC -> XS
      gemm(XN, M, C, C, bp.self_w, 0, bp.self_b, C, nullptr, nullptr, XS, 0);
      // pooled channel weights + tiny MLP + pair softmax
      pool_mean_k<<<cdiv((size_t)Bn * C, 256), 256, 0, stream>>>(XC, XS, Wp, Bn, HW, C);
      channel_mlp_k<<<Bn, 256, 0, stream>>>(Wp, bp.rw1_w, bp.rw1_b, bp.rw2_w, bp.rw2_b,
                                            Aw, Sw, C);
      // gated combine (in-place XC), then proj + residual -> X
      combine_k<<<cdiv((size_t)M * C, 256), 256, 0, stream>>>(XC, XS, Aw, Sw, HW, C,
                                                              (size_t)M * C);
      gemm(XC, M, C, C, bp.proj_w, 0, bp.proj_b, C, X, nullptr, X, 0);
      // LN2 -> XN (=y2); MLP: gelu(y2@m1) -> BIG; BIG@m2 + b + y2 + x -> X
      layernorm_k<<<cdiv(M, 8), 256, 0, stream>>>(X, bp.ln2_g, bp.ln2_b, XN, M, C);
      int Ch = 4 * C;
      gemm(XN, M, C, C, bp.m1_w, 0, bp.m1_b, Ch, nullptr, nullptr, BIG, 1);
      gemm(BIG, M, Ch, Ch, bp.m2_w, 0, bp.m2_b, C, XN, X, X, 0);
    }
    if (s < 3) {
      const int Cn = DIMS[s + 1];
      const int Ho = (H - 1) / 2 + 1;
      const int Mo = Bn * Ho * Ho;
      const int Kc = C * 9;
      im2col3x3_k<<<cdiv((size_t)Mo * Kc, 256), 256, 0, stream>>>(X, BIG, Bn, H, H, C,
                                                                  Ho, Ho);
      gemm(BIG, Mo, Kc, Kc, down_w[s], 1, down_b[s], Cn, nullptr, nullptr, XN, 0);
      float* tmp = X; X = XN; XN = tmp;   // ping-pong
      H = Ho;
    }
  }

  // ---- final LN, global pool, head (WMMA, M=32, N=1000 guarded)
  {
    const int C = 512, HW = H * H, M = Bn * HW;
    layernorm_k<<<cdiv(M, 8), 256, 0, stream>>>(X, norm_g, norm_b, XN, M, C);
    pool_mean_k<<<cdiv((size_t)Bn * C, 256), 256, 0, stream>>>(XN, nullptr, Wp, Bn, HW, C);
    gemm(Wp, Bn, C, C, head_w, 0, head_b, 1000, nullptr, nullptr, (float*)d_out, 0);
  }
}
